// SpatialGcnBone_11708080849105
// MI455X (gfx1250) — matrix operand order repaired and checked by reading, not compiled, not run on hardware
//
#include <hip/hip_runtime.h>
#include <hip/hip_bf16.h>
#include <math.h>

typedef _Float16 v16h __attribute__((ext_vector_type(16)));
typedef float    v8f  __attribute__((ext_vector_type(8)));
typedef _Float16 h2   __attribute__((ext_vector_type(2)));
typedef _Float16 h4   __attribute__((ext_vector_type(4)));
typedef _Float16 h8   __attribute__((ext_vector_type(8)));

#define TT    8                 // t-slab per block (300 = 37*8 + 4)
#define NCOL  (TT * 25)         // 200 columns per block
#define ROWP  264               // M row pitch in halves (16B-aligned rows)
#define XROWP 264               // sX row pitch in halves (256 data + 8 pad)
#define RROWP 40                // sR row pitch in halves
#define SM_M  (NCOL * ROWP)     // 52,800 halves
#define SM_X  (64 * XROWP)      // 16,896 halves
#define SM_R  (3 * 32 * RROWP)  //  3,840 halves
#define SMEM_HALVES (SM_M + SM_X + SM_R)
#define SMEM_BYTES  (SMEM_HALVES * 2)   // 147,072 B

// ---------------- Kernel 1: temporal mean pool: xt[n][c][v] ----------------
__global__ void pool_kernel(const float* __restrict__ x, float* __restrict__ xt) {
    const int b   = blockIdx.x;          // n*64 + c
    const int tid = threadIdx.x;         // 100 threads: (v, g)
    const int v = tid % 25, g = tid / 25;
    const float* base = x + (size_t)b * 7500;
    float s = 0.f;
    for (int t = g; t < 300; t += 4) s += base[t * 25 + v];
    __shared__ float part[100];
    part[tid] = s;
    __syncthreads();
    if (tid < 25) {
        float tot = part[tid] + part[tid + 25] + part[tid + 50] + part[tid + 75];
        xt[b * 25 + tid] = tot * (1.0f / 300.0f);
    }
}

// ---------------- Kernel 2: attention -> R, stored transposed+padded f16 ---
// Rt[n][s][w][v] (32x32, zero padded), Rt[w][v] = tanh(A1[:,v].A2[:,w]/16)+A[s,v,w]
__global__ void r_kernel(const float* __restrict__ xt, const float* __restrict__ A,
                         const float* __restrict__ wa0, const float* __restrict__ ba0,
                         const float* __restrict__ wa1, const float* __restrict__ ba1,
                         const float* __restrict__ wa2, const float* __restrict__ ba2,
                         _Float16* __restrict__ Rt) {
    const int n = blockIdx.x, s = blockIdx.y;
    const int tid = threadIdx.x;         // 128
    __shared__ float xtl[64 * 25];
    __shared__ float pooled[16 * 25], a1[16 * 25], a2[16 * 25];
    for (int e = tid; e < 1600; e += 128) xtl[e] = xt[n * 1600 + e];
    __syncthreads();
    for (int e = tid; e < 400; e += 128) {
        int i = e / 25, v = e % 25;
        float acc = ba0[s * 16 + i];
        for (int c = 0; c < 64; ++c) acc += wa0[(s * 16 + i) * 64 + c] * xtl[c * 25 + v];
        pooled[e] = acc;
    }
    __syncthreads();
    for (int e = tid; e < 800; e += 128) {
        int which = e / 400, e2 = e % 400;
        int j = e2 / 25, v = e2 % 25;
        const float* wa = which ? wa2 : wa1;
        const float* ba = which ? ba2 : ba1;
        float acc = ba[s * 16 + j];
        for (int i = 0; i < 16; ++i) acc += wa[(s * 16 + j) * 16 + i] * pooled[i * 25 + v];
        (which ? a2 : a1)[e2] = acc;
    }
    __syncthreads();
    _Float16* rdst = Rt + (size_t)(n * 3 + s) * 1024;
    for (int e = tid; e < 1024; e += 128) {
        int w = e / 32, v = e % 32;
        float val = 0.f;
        if (v < 25 && w < 25) {
            float acc = 0.f;
            for (int i = 0; i < 16; ++i) acc += a1[i * 25 + v] * a2[i * 25 + w];
            val = tanhf(acc * (1.0f / 16.0f)) + A[(s * 25 + v) * 25 + w];
        }
        rdst[w * 32 + v] = (_Float16)val;
    }
}

// ---------------- Kernel 3: fold BN into weights: W[o][k=256] f16, bias ----
__global__ void wbuild_kernel(const float* __restrict__ wp, const float* __restrict__ bp,
                              const float* __restrict__ wd,
                              const float* __restrict__ gamma, const float* __restrict__ beta,
                              const float* __restrict__ gamma_d, const float* __restrict__ beta_d,
                              _Float16* __restrict__ W, float* __restrict__ bias) {
    const int idx = blockIdx.x * 256 + threadIdx.x;  // 32768 total
    const int o = idx >> 8, k = idx & 255;
    const float inv = rsqrtf(1.0f + 1e-5f);
    const float gs = gamma[o] * inv, gd = gamma_d[o] * inv;
    const int s = k >> 6, c = k & 63;
    float val = (s < 3) ? wp[(s * 128 + o) * 64 + c] * gs : wd[o * 64 + c] * gd;
    W[o * 256 + k] = (_Float16)val;
    if (k == 0)
        bias[o] = (bp[o] + bp[128 + o] + bp[256 + o]) * gs + beta[o] + beta_d[o];
}

// ---------------- Kernel 4: fused XR + main GEMM + bias + ReLU -------------
__global__ void __launch_bounds__(256)
fused_kernel(const float* __restrict__ x, const _Float16* __restrict__ Rt,
             const _Float16* __restrict__ W, const float* __restrict__ bias,
             float* __restrict__ out) {
    extern __shared__ _Float16 smem[];
    _Float16* sM = smem;                 // [NCOL][ROWP] : col = t*25+v, k = 0..255
    _Float16* sX = smem + SM_M;          // [64 c][8 t * 32 vpad] (+8 pad/row)
    _Float16* sR = smem + SM_M + SM_X;   // [3 s][32 w][RROWP] (R^T: [w][v])

    const int n  = blockIdx.y;
    const int t0 = blockIdx.x * TT;
    const int tid   = threadIdx.x;
    const int wv    = tid >> 5;
    const int lane  = tid & 31;
    const int lhalf = lane & 15;
    const int hi    = lane >> 4;
    const int hi8   = hi << 3;

    // ---- stage x slab into M rows k=192..255 AND padded sX (float4 loads) ----
    {
        const float* xbase = x + (size_t)n * 64 * 7500 + (size_t)t0 * 25;
        const int lim = 7500 - t0 * 25;                     // valid cols
        for (int i4 = tid; i4 < 64 * (NCOL / 4); i4 += 256) {  // 3200 float4's
            int c = i4 / (NCOL / 4);
            int q = (i4 - c * (NCOL / 4)) * 4;
            float4 p = {0.f, 0.f, 0.f, 0.f};
            if (q < lim) p = *(const float4*)(xbase + (size_t)c * 7500 + q);
            _Float16 e[4] = {(_Float16)p.x, (_Float16)p.y, (_Float16)p.z, (_Float16)p.w};
            _Float16* dm = sM + 192 + c;
            _Float16* dx = sX + c * XROWP;
            int t = q / 25;
            int v = q - t * 25;
#pragma unroll
            for (int j = 0; j < 4; ++j) {
                dm[(q + j) * ROWP] = e[j];
                int vj = v + j;
                int tj = t, vv = vj;
                if (vj >= 25) { tj = t + 1; vv = vj - 25; }
                dx[tj * 32 + vv] = e[j];
            }
        }
        // zero-fill sX v = 25..31 (keeps WMMA K-padding exact / NaN-free)
        for (int j = tid; j < 64 * TT; j += 256) {
            int c = j >> 3, t = j & 7;
            _Float16* dz = sX + c * XROWP + t * 32;
            dz[25] = (_Float16)0.0f;
            *(h2*)(dz + 26) = (h2){(_Float16)0.0f, (_Float16)0.0f};
            *(h4*)(dz + 28) = (h4){(_Float16)0.0f, (_Float16)0.0f,
                                   (_Float16)0.0f, (_Float16)0.0f};
        }
    }
    // ---- stage R with repadded pitch: sR[s][w][RROWP] ----
    {
        const _Float16* rsrc = Rt + (size_t)n * 3072;
        for (int idx = tid; idx < 3072; idx += 256) {
            int v = idx & 31, w = (idx >> 5) & 31, s = idx >> 10;
            sR[(s * 32 + w) * RROWP + v] = rsrc[idx];
        }
    }
    __syncthreads();

    // ---- XR = X @ R[s] via WMMA, write into M rows k = s*64+c ----
    // t-major rows: row = t*64 + c ; tile rt: t = rt>>2, c = ((rt&3)<<4)+m
    for (int job = wv; job < 3 * 32 * 2; job += 8) {
        const int s   = job >> 6;
        const int rem = job & 63;
        const int rt  = rem >> 1;
        const int wt  = rem & 1;
        const int t   = rt >> 2;
        const int c0  = (rt & 3) << 4;
        // A fragment: rows m = lhalf -> c = c0+m, fixed t ; two b128 loads
        const _Float16* ax = sX + (c0 + lhalf) * XROWP + t * 32 + hi8;
        h8 a0 = *(const h8*)(ax);
        h8 a1 = *(const h8*)(ax + 16);
        v16h a = __builtin_shufflevector(a0, a1, 0, 1, 2, 3, 4, 5, 6, 7,
                                         8, 9, 10, 11, 12, 13, 14, 15);
        // B fragment: R[v][w] from sR[s][w][v] ; two b128 loads
        const int w = wt * 16 + lhalf;
        const _Float16* rb = sR + (s * 32 + w) * RROWP + hi8;
        h8 b0 = *(const h8*)(rb);
        h8 b1 = *(const h8*)(rb + 16);
        v16h b = __builtin_shufflevector(b0, b1, 0, 1, 2, 3, 4, 5, 6, 7,
                                         8, 9, 10, 11, 12, 13, 14, 15);
        v8f acc = {};
        acc = __builtin_amdgcn_wmma_f32_16x16x32_f16(false, a, false, b,
                                                     (short)0, acc, false, false);
        // D: rows m = r + 8*hi -> c = c0 + r + 8*hi (k-contiguous) ; one b128
        if (w < 25) {
            _Float16* dst = sM + (size_t)(t * 25 + w) * ROWP + s * 64 + c0 + hi8;
            h8 d;
#pragma unroll
            for (int r = 0; r < 8; ++r) d[r] = (_Float16)acc[r];
            *(h8*)(dst) = d;
        }
    }
    __syncthreads();

    // ---- main GEMM: out[o, col] = W[o, 0:256] . M[0:256, col] ----
    v16h aw[8];
    {
        const _Float16* wbase = W + (size_t)(16 * wv + lhalf) * 256;
#pragma unroll
        for (int ks = 0; ks < 8; ++ks) {
#pragma unroll
            for (int r = 0; r < 8; ++r) {
                int k = 32 * ks + ((r < 4) ? (2 * r + hi8)
                                           : (16 + 2 * (r - 4) + hi8));
                h2 p = *(const h2*)(wbase + k);
                aw[ks][2 * r] = p.x; aw[ks][2 * r + 1] = p.y;
            }
        }
    }
    float bvals[8];
#pragma unroll
    for (int r = 0; r < 8; ++r) bvals[r] = bias[16 * wv + r + hi8];

    for (int ct = 0; ct < 13; ++ct) {                 // ceil(200/16) col tiles
        const int col = 16 * ct + lhalf;              // tail lanes read junk,
        v8f acc = {};                                 // store is guarded below
        const _Float16* mcol = sM + (size_t)col * ROWP + hi8;
#pragma unroll
        for (int ks = 0; ks < 8; ++ks) {
            h8 b0 = *(const h8*)(mcol + 32 * ks);
            h8 b1 = *(const h8*)(mcol + 32 * ks + 16);
            v16h b = __builtin_shufflevector(b0, b1, 0, 1, 2, 3, 4, 5, 6, 7,
                                             8, 9, 10, 11, 12, 13, 14, 15);
            acc = __builtin_amdgcn_wmma_f32_16x16x32_f16(false, aw[ks], false, b,
                                                         (short)0, acc, false, false);
        }
        const int tl = col / 25, v = col - tl * 25;
        const int t  = t0 + tl;
        if (col < NCOL && t < 300) {
            float* ob = out + ((size_t)(n * 128 + 16 * wv + hi8) * 300 + t) * 25 + v;
#pragma unroll
            for (int r = 0; r < 8; ++r) {
                float val = acc[r] + bvals[r];
                ob[(size_t)r * 7500] = fmaxf(val, 0.0f);
            }
        }
    }
}

// ---------------------------------------------------------------------------
extern "C" void kernel_launch(void* const* d_in, const int* in_sizes, int n_in,
                              void* d_out, int out_size, void* d_ws, size_t ws_size,
                              hipStream_t stream) {
    const float* x       = (const float*)d_in[0];
    const float* A       = (const float*)d_in[1];
    const float* wp      = (const float*)d_in[2];
    const float* bp      = (const float*)d_in[3];
    const float* wa0     = (const float*)d_in[4];
    const float* ba0     = (const float*)d_in[5];
    const float* wa1     = (const float*)d_in[6];
    const float* ba1     = (const float*)d_in[7];
    const float* wa2     = (const float*)d_in[8];
    const float* ba2     = (const float*)d_in[9];
    const float* wd      = (const float*)d_in[10];
    const float* gamma   = (const float*)d_in[11];
    const float* beta    = (const float*)d_in[12];
    const float* gamma_d = (const float*)d_in[13];
    const float* beta_d  = (const float*)d_in[14];
    float* out = (float*)d_out;

    char* ws = (char*)d_ws;
    float*     xt   = (float*)ws;                  // 64*64*25 f32   = 409,600 B
    _Float16*  Rt   = (_Float16*)(ws + 409600);    // 64*3*32*32 f16 = 393,216 B
    _Float16*  Wm   = (_Float16*)(ws + 802816);    // 128*256 f16    =  65,536 B
    float*     bias = (float*)(ws + 868352);       // 128 f32        =     512 B

    pool_kernel<<<64 * 64, 100, 0, stream>>>(x, xt);
    r_kernel<<<dim3(64, 3), 128, 0, stream>>>(xt, A, wa0, ba0, wa1, ba1, wa2, ba2, Rt);
    wbuild_kernel<<<128, 256, 0, stream>>>(wp, bp, wd, gamma, beta, gamma_d, beta_d, Wm, bias);
    fused_kernel<<<dim3(38, 64), 256, SMEM_BYTES, stream>>>(x, Rt, Wm, bias, out);
}
